// DESMHSA_35235911696962
// MI455X (gfx1250) — matrix operand order repaired and checked by reading, not compiled, not run on hardware
//
#include <hip/hip_runtime.h>

// Problem constants (match reference)
#define DIMC 512
#define NHEAD 8
#define NFREQ 8
#define BSZ 2
#define NTOK 1024
#define HDIM 64

typedef __attribute__((ext_vector_type(16))) _Float16 v16h;
typedef __attribute__((ext_vector_type(8)))  _Float16 v8h;
typedef __attribute__((ext_vector_type(8)))  float    v8f;

union AFrag { v16h v; v8h h[2]; };

static __device__ __forceinline__ float softplus_f(float x) {
    return (x > 20.f) ? x : log1pf(__expf(x));
}
static __device__ __forceinline__ float rmax16(float v) {
    v = fmaxf(v, __shfl_xor(v, 1));
    v = fmaxf(v, __shfl_xor(v, 2));
    v = fmaxf(v, __shfl_xor(v, 4));
    v = fmaxf(v, __shfl_xor(v, 8));
    return v;
}
static __device__ __forceinline__ float rsum16(float v) {
    v += __shfl_xor(v, 1);
    v += __shfl_xor(v, 2);
    v += __shfl_xor(v, 4);
    v += __shfl_xor(v, 8);
    return v;
}

// ---------------- f32 -> f16 convert ----------------
__global__ void convert_f16_kernel(_Float16* __restrict__ dst,
                                   const float* __restrict__ src, int n) {
    for (int i = blockIdx.x * blockDim.x + threadIdx.x; i < n;
         i += gridDim.x * blockDim.x)
        dst[i] = (_Float16)src[i];
}

// Transpose (rows x cols) f32 -> (cols x rows) f16. Coalesced writes.
__global__ void transpose_f16_kernel(_Float16* __restrict__ dst,
                                     const float* __restrict__ src,
                                     int rows, int cols) {
    int n = rows * cols;
    for (int idx = blockIdx.x * blockDim.x + threadIdx.x; idx < n;
         idx += gridDim.x * blockDim.x) {
        int o = idx / rows;   // output row  (= input col)
        int i = idx % rows;   // output col  (= input row)
        dst[idx] = (_Float16)src[i * cols + o];
    }
}

// ---------------- QKV projection GEMM: (2048x512) @ (512x1536) ----------------
// One wave per 16x16 output tile. Epilogue scatters into per-head layouts:
//   Q, K : [B,H,N,HD] row-major ; V stored transposed: [B,H,HD,N]
__global__ __launch_bounds__(128) void qkv_gemm_kernel(
    const _Float16* __restrict__ X, const _Float16* __restrict__ WT,
    const float* __restrict__ bias, _Float16* __restrict__ Q,
    _Float16* __restrict__ K, _Float16* __restrict__ VT) {
    const int lane = threadIdx.x & 31, warp = threadIdx.x >> 5;
    const int tile = blockIdx.x * 4 + warp;
    const int ct = tile % (3 * DIMC / 16);   // 96 col tiles
    const int rt = tile / (3 * DIMC / 16);   // 128 row tiles
    const int lr = lane & 15, hi = lane >> 4;

    v8f acc = {};
    const _Float16* arow = X + (size_t)(rt * 16 + lr) * DIMC;
    const _Float16* brow = WT + (size_t)(ct * 16 + lr) * DIMC;
    for (int k = 0; k < DIMC; k += 32) {
        AFrag a;
        a.h[0] = *(const v8h*)(arow + k + hi * 16);       // A: K 0..7 / 16..23
        a.h[1] = *(const v8h*)(arow + k + 8 + hi * 16);   //    K 8..15 / 24..31
        v16h bf = *(const v16h*)(brow + k + hi * 16);     // B: K 0..15 / 16..31
        acc = __builtin_amdgcn_wmma_f32_16x16x32_f16(false, a.v, false, bf,
                                                     (short)0, acc, false, false);
    }
    const int c = ct * 16 + lr;
    const float bc = bias[c];
    for (int r = 0; r < 8; ++r) {
        int row = rt * 16 + r + 8 * hi;          // C/D layout: M = r + 8*(lane>=16)
        int b = row >> 10, n = row & (NTOK - 1);
        _Float16 hv = (_Float16)(acc[r] + bc);
        if (c < DIMC) {
            int h = c >> 6, hd = c & 63;
            Q[(((size_t)(b * NHEAD + h)) * NTOK + n) * HDIM + hd] = hv;
        } else if (c < 2 * DIMC) {
            int cc = c - DIMC, h = cc >> 6, hd = cc & 63;
            K[(((size_t)(b * NHEAD + h)) * NTOK + n) * HDIM + hd] = hv;
        } else {
            int cc = c - 2 * DIMC, h = cc >> 6, hd = cc & 63;
            VT[(((size_t)(b * NHEAD + h)) * HDIM + hd) * NTOK + n] = hv;
        }
    }
}

// ---------------- Fused flash attention with distance-encoded bias ----------
// One wave per (b,h, 16-query tile). Streams 32 keys per iteration:
// 4 WMMAs QK^T, inline bias + online softmax, P staged through wave-private
// LDS (C-layout -> A-layout transpose), 4 WMMAs P@V with running rescale.
__global__ __launch_bounds__(128) void flash_attn_kernel(
    const _Float16* __restrict__ Q, const _Float16* __restrict__ K,
    const _Float16* __restrict__ VT, const float* __restrict__ coords,
    const float* __restrict__ omega_raw, const float* __restrict__ a_p,
    const float* __restrict__ c_p, const float* __restrict__ alpha_raw,
    const float* __restrict__ ell_raw, const float* __restrict__ bs_raw,
    _Float16* __restrict__ O) {
    __shared__ _Float16 ldsP[4][16 * 32];     // 1KB per wave
    const int lane = threadIdx.x & 31, warp = threadIdx.x >> 5;
    const int wid = blockIdx.x * 4 + warp;
    const int qt = wid & 63;                  // 64 query tiles per (b,h)
    const int bh = wid >> 6;
    const int b = bh >> 3, h = bh & 7;
    const int lr = lane & 15, hi = lane >> 4;

    // per-head parameters (broadcast into every lane)
    const float alpha = softplus_f(alpha_raw[h]) + 1e-12f;
    const float ell = softplus_f(ell_raw[h]) + 1e-12f;
    const float inv_ell2 = 1.f / (ell * ell);
    const float tbs = tanhf(bs_raw[h]);
    float om[NFREQ], af[NFREQ], cf[NFREQ];
    for (int f = 0; f < NFREQ; ++f) {
        om[f] = softplus_f(omega_raw[h * NFREQ + f]) + 1e-12f;
        af[f] = a_p[h * NFREQ + f];
        cf[f] = c_p[h * NFREQ + f];
    }

    // Q A-operands for hd 0..31 and 32..63 (loaded once)
    const _Float16* qrow = Q + ((size_t)bh * NTOK + qt * 16 + lr) * HDIM;
    AFrag qa0, qa1;
    qa0.h[0] = *(const v8h*)(qrow + hi * 16);
    qa0.h[1] = *(const v8h*)(qrow + 8 + hi * 16);
    qa1.h[0] = *(const v8h*)(qrow + 32 + hi * 16);
    qa1.h[1] = *(const v8h*)(qrow + 40 + hi * 16);

    float cq[8];
    for (int r = 0; r < 8; ++r) cq[r] = coords[b * NTOK + qt * 16 + r + 8 * hi];

    float mrow[8], lsum[8];
    for (int r = 0; r < 8; ++r) { mrow[r] = -1e30f; lsum[r] = 0.f; }
    v8f o0 = {}, o1 = {}, o2 = {}, o3 = {};
    _Float16* pw = ldsP[warp];
    const float scale = 0.125f;               // HD^-0.5

    for (int kb = 0; kb < NTOK / 32; ++kb) {
        // --- scores: two 16x16 tiles over 32 keys ---
        v8f s0 = {}, s1 = {};
        const _Float16* krow0 = K + ((size_t)bh * NTOK + kb * 32 + lr) * HDIM;
        const _Float16* krow1 = krow0 + 16 * HDIM;
        {
            v16h k00 = *(const v16h*)(krow0 + hi * 16);
            v16h k01 = *(const v16h*)(krow0 + 32 + hi * 16);
            s0 = __builtin_amdgcn_wmma_f32_16x16x32_f16(false, qa0.v, false, k00,
                                                        (short)0, s0, false, false);
            s0 = __builtin_amdgcn_wmma_f32_16x16x32_f16(false, qa1.v, false, k01,
                                                        (short)0, s0, false, false);
            v16h k10 = *(const v16h*)(krow1 + hi * 16);
            v16h k11 = *(const v16h*)(krow1 + 32 + hi * 16);
            s1 = __builtin_amdgcn_wmma_f32_16x16x32_f16(false, qa0.v, false, k10,
                                                        (short)0, s1, false, false);
            s1 = __builtin_amdgcn_wmma_f32_16x16x32_f16(false, qa1.v, false, k11,
                                                        (short)0, s1, false, false);
        }
        const float ck0 = coords[b * NTOK + kb * 32 + lr];
        const float ck1 = coords[b * NTOK + kb * 32 + 16 + lr];

        // --- bias + online softmax per C/D row r ---
        for (int r = 0; r < 8; ++r) {
            float d0 = fabsf(cq[r] - ck0), d1 = fabsf(cq[r] - ck1);
            float b0 = alpha * __expf(-d0 * d0 * inv_ell2);
            float b1 = alpha * __expf(-d1 * d1 * inv_ell2);
            for (int f = 0; f < NFREQ; ++f) {
                b0 += af[f] * __cosf(d0 * om[f]) + cf[f] * __sinf(d0 * om[f]);
                b1 += af[f] * __cosf(d1 * om[f]) + cf[f] * __sinf(d1 * om[f]);
            }
            float v0 = s0[r] * scale + tbs * b0;
            float v1 = s1[r] * scale + tbs * b1;
            float mn = fmaxf(mrow[r], rmax16(fmaxf(v0, v1)));
            float corr = __expf(mrow[r] - mn);
            mrow[r] = mn;
            float p0 = __expf(v0 - mn), p1 = __expf(v1 - mn);
            lsum[r] = lsum[r] * corr + rsum16(p0 + p1);
            o0[r] *= corr; o1[r] *= corr; o2[r] *= corr; o3[r] *= corr;
            int row = r + 8 * hi;
            pw[row * 32 + lr] = (_Float16)p0;
            pw[row * 32 + 16 + lr] = (_Float16)p1;
        }
        // wave-private LDS: DS ops are in-order; just drain before re-reading
        asm volatile("s_wait_dscnt 0x0" ::: "memory");

        // P in A-operand layout (16x32, K = 32 keys)
        AFrag pa;
        pa.h[0] = *(const v8h*)(pw + lr * 32 + hi * 16);
        pa.h[1] = *(const v8h*)(pw + lr * 32 + 8 + hi * 16);

        // V B-operands: VT[b,h,hd,N] -> per lane contiguous 16 keys
        const _Float16* vbase = VT + ((size_t)bh * HDIM) * NTOK + kb * 32 + hi * 16;
        v16h vb0 = *(const v16h*)(vbase + (size_t)(0 * 16 + lr) * NTOK);
        v16h vb1 = *(const v16h*)(vbase + (size_t)(1 * 16 + lr) * NTOK);
        v16h vb2 = *(const v16h*)(vbase + (size_t)(2 * 16 + lr) * NTOK);
        v16h vb3 = *(const v16h*)(vbase + (size_t)(3 * 16 + lr) * NTOK);
        o0 = __builtin_amdgcn_wmma_f32_16x16x32_f16(false, pa.v, false, vb0,
                                                    (short)0, o0, false, false);
        o1 = __builtin_amdgcn_wmma_f32_16x16x32_f16(false, pa.v, false, vb1,
                                                    (short)0, o1, false, false);
        o2 = __builtin_amdgcn_wmma_f32_16x16x32_f16(false, pa.v, false, vb2,
                                                    (short)0, o2, false, false);
        o3 = __builtin_amdgcn_wmma_f32_16x16x32_f16(false, pa.v, false, vb3,
                                                    (short)0, o3, false, false);
    }

    // normalize + store O as [B,N,H*HD] f16 (A operand of proj GEMM)
    for (int r = 0; r < 8; ++r) {
        float inv = 1.f / lsum[r];
        int row = qt * 16 + r + 8 * hi;
        _Float16* ob = O + ((size_t)(b * NTOK + row)) * DIMC + h * HDIM + lr;
        ob[0]  = (_Float16)(o0[r] * inv);
        ob[16] = (_Float16)(o1[r] * inv);
        ob[32] = (_Float16)(o2[r] * inv);
        ob[48] = (_Float16)(o3[r] * inv);
    }
}

// ---------------- Output projection GEMM: (2048x512) @ (512x512) + b --------
__global__ __launch_bounds__(128) void proj_gemm_kernel(
    const _Float16* __restrict__ A, const _Float16* __restrict__ WT,
    const float* __restrict__ bias, float* __restrict__ out) {
    const int lane = threadIdx.x & 31, warp = threadIdx.x >> 5;
    const int tile = blockIdx.x * 4 + warp;
    const int ct = tile & 31;     // 32 col tiles
    const int rt = tile >> 5;     // 128 row tiles
    const int lr = lane & 15, hi = lane >> 4;

    v8f acc = {};
    const _Float16* arow = A + (size_t)(rt * 16 + lr) * DIMC;
    const _Float16* brow = WT + (size_t)(ct * 16 + lr) * DIMC;
    for (int k = 0; k < DIMC; k += 32) {
        AFrag a;
        a.h[0] = *(const v8h*)(arow + k + hi * 16);
        a.h[1] = *(const v8h*)(arow + k + 8 + hi * 16);
        v16h bf = *(const v16h*)(brow + k + hi * 16);
        acc = __builtin_amdgcn_wmma_f32_16x16x32_f16(false, a.v, false, bf,
                                                     (short)0, acc, false, false);
    }
    const int c = ct * 16 + lr;
    const float bc = bias[c];
    for (int r = 0; r < 8; ++r) {
        int row = rt * 16 + r + 8 * hi;
        out[(size_t)row * DIMC + c] = acc[r] + bc;
    }
}

extern "C" void kernel_launch(void* const* d_in, const int* in_sizes, int n_in,
                              void* d_out, int out_size, void* d_ws, size_t ws_size,
                              hipStream_t stream) {
    const float* x         = (const float*)d_in[0];
    const float* coords    = (const float*)d_in[1];
    const float* qkv_w     = (const float*)d_in[2];
    const float* qkv_b     = (const float*)d_in[3];
    const float* proj_w    = (const float*)d_in[4];
    const float* proj_b    = (const float*)d_in[5];
    const float* omega_raw = (const float*)d_in[6];
    const float* a_p       = (const float*)d_in[7];
    const float* c_p       = (const float*)d_in[8];
    const float* alpha_raw = (const float*)d_in[9];
    const float* ell_raw   = (const float*)d_in[10];
    const float* bs_raw    = (const float*)d_in[11];

    char* ws = (char*)d_ws;
    _Float16* X16  = (_Float16*)ws; ws += (size_t)BSZ * NTOK * DIMC * 2;
    _Float16* WT16 = (_Float16*)ws; ws += (size_t)DIMC * 3 * DIMC * 2;
    _Float16* PT16 = (_Float16*)ws; ws += (size_t)DIMC * DIMC * 2;
    _Float16* Q16  = (_Float16*)ws; ws += (size_t)BSZ * NHEAD * NTOK * HDIM * 2;
    _Float16* K16  = (_Float16*)ws; ws += (size_t)BSZ * NHEAD * NTOK * HDIM * 2;
    _Float16* VT16 = (_Float16*)ws; ws += (size_t)BSZ * NHEAD * NTOK * HDIM * 2;
    _Float16* O16  = (_Float16*)ws; ws += (size_t)BSZ * NTOK * DIMC * 2;
    (void)ws_size; (void)in_sizes; (void)n_in; (void)out_size;

    const int nx = BSZ * NTOK * DIMC;
    convert_f16_kernel<<<1024, 256, 0, stream>>>(X16, x, nx);
    transpose_f16_kernel<<<(DIMC * 3 * DIMC + 255) / 256, 256, 0, stream>>>(
        WT16, qkv_w, DIMC, 3 * DIMC);
    transpose_f16_kernel<<<(DIMC * DIMC + 255) / 256, 256, 0, stream>>>(
        PT16, proj_w, DIMC, DIMC);

    // 128 row tiles * 96 col tiles, 4 waves/block
    qkv_gemm_kernel<<<(128 * 96) / 4, 128, 0, stream>>>(X16, WT16, qkv_b,
                                                        Q16, K16, VT16);
    // B*H*64 q-tiles = 1024 waves, 4 waves/block
    flash_attn_kernel<<<256, 128, 0, stream>>>(Q16, K16, VT16, coords, omega_raw,
                                               a_p, c_p, alpha_raw, ell_raw,
                                               bs_raw, O16);
    // 128 row tiles * 32 col tiles, 4 waves/block
    proj_gemm_kernel<<<(128 * 32) / 4, 128, 0, stream>>>(O16, PT16, proj_b,
                                                         (float*)d_out);
}
